// SelfAttentionForHW_87376814669856
// MI455X (gfx1250) — matrix operand order repaired and checked by reading, not compiled, not run on hardware
//
#include <hip/hip_runtime.h>
#include <hip/hip_bf16.h>
#include <math.h>

typedef __attribute__((ext_vector_type(2)))  float  v2f;
typedef __attribute__((ext_vector_type(4)))  float  v4f;
typedef __attribute__((ext_vector_type(8)))  float  v8f;
typedef __attribute__((ext_vector_type(4)))  __bf16 v4bf;
typedef __attribute__((ext_vector_type(8)))  __bf16 v8bf;
typedef __attribute__((ext_vector_type(16))) __bf16 v16bf;
typedef __attribute__((ext_vector_type(4))) unsigned int u32x4;
typedef __attribute__((ext_vector_type(4))) int          i32x4;
typedef __attribute__((ext_vector_type(8))) int          i32x8;

#define BATCH 4
#define CCH   64
#define TT    8
#define NPOS  4096   // H*W
#define DQK   64     // (C/8)*T
#define DV    512    // C*T

#if __has_include(<hip/amd_detail/amd_gfx1250_TDM.h>)
#define TDM_6ARG 1
#else
#define TDM_6ARG 0
#endif

// ---------------------------------------------------------------------------
// TDM: issue a 2D tensor_load_to_lds.  D# packing per CDNA5 ISA §8 (groups
// 0/1; groups 2/3 zero => 2D).  data_size code: 0=1B,1=2B,2=4B,3=8B.
// pad: after every (1<<pad_int) 8-byte units, add (pad_amt+1) dwords to LDS.
// ---------------------------------------------------------------------------
__device__ __forceinline__ void tdm_load_2d(
    unsigned lds_off, const void* gptr,
    unsigned ds_code,
    unsigned tensor_d0, unsigned tensor_d1,
    unsigned tile_d0, unsigned tile_d1,
    unsigned d0_stride,
    unsigned pad_en, unsigned pad_int, unsigned pad_amt)
{
    unsigned long long ga = (unsigned long long)gptr;
    u32x4 g0;
    g0[0] = 1u;                                            // count=1 (user D#)
    g0[1] = lds_off;                                       // lds_addr
    g0[2] = (unsigned)(ga & 0xffffffffu);                  // global_addr[31:0]
    g0[3] = (unsigned)((ga >> 32) & 0x01ffffffu) | (2u << 30);  // addr[56:32] | type=2

    i32x8 g1;
    g1[0] = (int)((ds_code << 16) | (pad_en << 20) | (pad_int << 22) | (pad_amt << 25));
    g1[1] = (int)((tensor_d0 & 0xffffu) << 16);            // tensor_dim0[15:0]
    g1[2] = (int)(((tensor_d0 >> 16) & 0xffffu) | ((tensor_d1 & 0xffffu) << 16));
    g1[3] = (int)(((tensor_d1 >> 16) & 0xffffu) | ((tile_d0 & 0xffffu) << 16));
    g1[4] = (int)(tile_d1 & 0xffffu);                      // tile_dim1, tile_dim2=0
    g1[5] = (int)d0_stride;                                // tensor_dim0_stride[31:0]
    g1[6] = 0;
    g1[7] = 0;

    i32x4 gz = {0, 0, 0, 0};
#if TDM_6ARG
    i32x8 gz8 = {0, 0, 0, 0, 0, 0, 0, 0};
    __builtin_amdgcn_tensor_load_to_lds(g0, g1, gz, gz, gz8, 0);
#else
    __builtin_amdgcn_tensor_load_to_lds(g0, g1, gz, gz, 0);
#endif
}

__device__ __forceinline__ unsigned lds_off_of(const void* p) {
    // generic shared-aperture address: low 32 bits are the LDS byte address
    return (unsigned)(unsigned long long)p;
}

// ---------------------------------------------------------------------------
// Kernel 1: 1x1x1 conv projections.
// q,k -> fp32 workspace in TRANSPOSED [b][n][d] layout (feeds ds_load_b64
// WMMA fragments in the energy kernel); v -> bf16 workspace [b][d][n].
// ---------------------------------------------------------------------------
__global__ __launch_bounds__(256) void proj_kernel(
    const float* __restrict__ x,
    const float* __restrict__ Wq, const float* __restrict__ bq,
    const float* __restrict__ Wk, const float* __restrict__ bk,
    const float* __restrict__ Wv, const float* __restrict__ bv,
    float* __restrict__ qws, float* __restrict__ kws, __bf16* __restrict__ vws)
{
    __shared__ float sWq[8 * 64], sWk[8 * 64], sWv[64 * 64];
    __shared__ float sbq[8], sbk[8], sbv[64];
    const int tid = threadIdx.x;
    for (int i = tid; i < 8 * 64; i += 256) { sWq[i] = Wq[i]; sWk[i] = Wk[i]; }
    for (int i = tid; i < 64 * 64; i += 256) sWv[i] = Wv[i];
    if (tid < 8)  { sbq[tid] = bq[tid]; sbk[tid] = bk[tid]; }
    if (tid < 64) sbv[tid] = bv[tid];
    __syncthreads();

    const long flat = (long)blockIdx.x * 256 + tid;  // b*T*N + t*N + n
    const int n  = (int)(flat % NPOS);
    const int bt = (int)(flat / NPOS);
    const int t  = bt % TT;
    const int b  = bt / TT;

    float xv[64];
    const float* xb = x + ((long)(b * CCH) * TT + t) * NPOS + n;
#pragma unroll
    for (int c = 0; c < 64; c++) xv[c] = xb[(long)c * TT * NPOS];

    float* qrow = qws + ((long)b * NPOS + n) * DQK;
    float* krow = kws + ((long)b * NPOS + n) * DQK;
#pragma unroll
    for (int o = 0; o < 8; o++) {
        float aq = sbq[o], ak = sbk[o];
#pragma unroll
        for (int c = 0; c < 64; c++) {
            aq = fmaf(sWq[o * 64 + c], xv[c], aq);
            ak = fmaf(sWk[o * 64 + c], xv[c], ak);
        }
        const int d = o * TT + t;
        qrow[d] = aq;
        krow[d] = ak;
    }
#pragma unroll 4
    for (int o = 0; o < 64; o++) {
        float av = sbv[o];
#pragma unroll
        for (int c = 0; c < 64; c++) av = fmaf(sWv[o * 64 + c], xv[c], av);
        const int d = o * TT + t;
        vws[(long)(b * DV + d) * NPOS + n] = (__bf16)av;
    }
}

// ---------------------------------------------------------------------------
// Kernel 2: energy = q^T k, fp32 WMMA 16x16x4, 128(n) x 64(m) tile per block.
// q/k tiles TDM-staged in [n][d] layout with +2-dword row padding (66-dword
// stride => conflict-free ds_load_b64 fragments straight into WMMA operands).
// ---------------------------------------------------------------------------
__global__ __launch_bounds__(256) void energy_kernel(
    const float* __restrict__ qws, const float* __restrict__ kws,
    float* __restrict__ att)
{
    __shared__ float qsT[128][66];  // [n][d]  33 KB (64 data + 2 pad dwords)
    __shared__ float ksT[64][66];   // [m][d]  16.5 KB
    const int b  = blockIdx.z;
    const int n0 = blockIdx.y * 128;
    const int m0 = blockIdx.x * 64;
    const int tid = threadIdx.x;

    const float* qb = qws + ((long)b * NPOS + n0) * DQK;
    const float* kb = kws + ((long)b * NPOS + m0) * DQK;

    if (tid < 32) {  // wave 0 drives the TDM (EXEC-independent, per-wave op)
        // rows of 64 f32 (256B); pad 2 dwords per 64 dwords -> 66-dword stride
        tdm_load_2d(lds_off_of(&qsT[0][0]), qb, /*ds=*/2,
                    DQK, NPOS - n0, /*tile*/64, 128, DQK,
                    /*pad_en=*/1, /*pad_int(64dw)=*/5, /*pad_amt(2dw)=*/1);
        tdm_load_2d(lds_off_of(&ksT[0][0]), kb, /*ds=*/2,
                    DQK, NPOS - m0, /*tile*/64, 64, DQK,
                    /*pad_en=*/1, /*pad_int(64dw)=*/5, /*pad_amt(2dw)=*/1);
        __builtin_amdgcn_s_wait_tensorcnt(0);
    }
    __syncthreads();

    const int lane = tid & 31, w = tid >> 5;
    const int lm = lane & 15, hi = lane >> 4;
    const int nb = (w >> 1) * 32;   // 4 wave-rows over n
    const int mb = (w & 1) * 32;    // 2 wave-cols over m

    v8f acc[2][2] = {};
#pragma unroll
    for (int kk = 0; kk < 16; kk++) {
        const int k2 = kk * 4 + 2 * hi;  // A/B 16x4 f32: VGPR0={K0|K2}, VGPR1={K1|K3}
        v2f a[2], bfrag[2];
#pragma unroll
        for (int i = 0; i < 2; i++)
            a[i] = *(const v2f*)&qsT[nb + i * 16 + lm][k2];
#pragma unroll
        for (int j = 0; j < 2; j++)
            bfrag[j] = *(const v2f*)&ksT[mb + j * 16 + lm][k2];
#pragma unroll
        for (int i = 0; i < 2; i++)
#pragma unroll
            for (int j = 0; j < 2; j++)
                acc[i][j] = __builtin_amdgcn_wmma_f32_16x16x4_f32(
                    false, a[i], false, bfrag[j], (short)0, acc[i][j], false, false);
    }

#pragma unroll
    for (int i = 0; i < 2; i++)
#pragma unroll
        for (int j = 0; j < 2; j++) {
            const int col = m0 + mb + j * 16 + lm;
#pragma unroll
            for (int r = 0; r < 8; r++) {
                const int row = n0 + nb + i * 16 + hi * 8 + r;
                att[((long)b * NPOS + row) * (long)NPOS + col] = acc[i][j][r];
            }
        }
}

// ---------------------------------------------------------------------------
// Kernel 3: in-place row softmax. One block per (b, n) row; row in registers.
// ---------------------------------------------------------------------------
__global__ __launch_bounds__(256) void softmax_kernel(float* __restrict__ att)
{
    __shared__ float red[256];
    const long row = blockIdx.x;
    float* p = att + row * (long)NPOS;
    const int tid = threadIdx.x;

    float v[16];
    float m = -3.4e38f;
#pragma unroll
    for (int j = 0; j < 16; j++) { v[j] = p[tid + j * 256]; m = fmaxf(m, v[j]); }
    red[tid] = m; __syncthreads();
    for (int s = 128; s > 0; s >>= 1) {
        if (tid < s) red[tid] = fmaxf(red[tid], red[tid + s]);
        __syncthreads();
    }
    const float rowmax = red[0];
    __syncthreads();

    float sum = 0.f;
#pragma unroll
    for (int j = 0; j < 16; j++) { v[j] = __expf(v[j] - rowmax); sum += v[j]; }
    red[tid] = sum; __syncthreads();
    for (int s = 128; s > 0; s >>= 1) {
        if (tid < s) red[tid] += red[tid + s];
        __syncthreads();
    }
    const float inv = 1.0f / red[0];
#pragma unroll
    for (int j = 0; j < 16; j++) p[tid + j * 256] = v[j] * inv;
}

// ---------------------------------------------------------------------------
// Kernel 4: out = gamma * (V · att^T) + x ; bf16 WMMA 16x16x32, K = 4096.
// Double-buffered software pipeline: TDM for tile i+1 and the att(i+1)
// b128+cvt stage are issued right after the barrier and overlap the WMMAs
// on tile i.  One barrier per iteration.
// ---------------------------------------------------------------------------
__global__ __launch_bounds__(256) void av_kernel(
    const __bf16* __restrict__ vws, const float* __restrict__ att,
    const float* __restrict__ x, const float* __restrict__ gammap,
    float* __restrict__ out)
{
    __shared__ __bf16 As[2][128][40];  // [d][k] 32 halfs + 8 pad (TDM pads rows)
    __shared__ __bf16 Bs[2][64][40];   // [n][k] 32 halfs + 8 pad (att, bf16)
    const int b  = blockIdx.z;
    const int d0 = blockIdx.y * 128;
    const int n0 = blockIdx.x * 64;
    const int tid = threadIdx.x;
    const int lane = tid & 31, w = tid >> 5;
    const int lm = lane & 15, hi = lane >> 4;
    const int db  = (w >> 1) * 32;  // 4 wave-rows over d
    const int nb2 = (w & 1) * 32;   // 2 wave-cols over n

    const __bf16* vb = vws + (long)(b * DV + d0) * NPOS;
    const float*  ab = att + (long)b * NPOS * NPOS + (long)n0 * NPOS;

    // prologue: tile 0 in flight
    if (tid < 32)
        tdm_load_2d(lds_off_of(&As[0][0][0]), vb, /*ds=*/1,
                    NPOS, 128, /*tile*/32, 128, NPOS,
                    /*pad_en=*/1, /*pad_int(16dw)=*/3, /*pad_amt(4dw)=*/3);
    for (int e = tid; e < 512; e += 256) {
        const int n = e >> 3, c4 = e & 7;
        v4f f = *(const v4f*)&ab[(long)n * NPOS + c4 * 4];
        v4bf bv;
        bv[0] = (__bf16)f.x; bv[1] = (__bf16)f.y;
        bv[2] = (__bf16)f.z; bv[3] = (__bf16)f.w;
        *(v4bf*)&Bs[0][n][c4 * 4] = bv;
    }

    v8f acc[2][2] = {};
    for (int it = 0; it < NPOS / 32; ++it) {
        const int cur = it & 1, nxt = cur ^ 1;
        const int m0 = it * 32;
        if (tid < 32) __builtin_amdgcn_s_wait_tensorcnt(0);  // As[cur] ready
        __syncthreads();                                     // Bs[cur] ready too

        if (it + 1 < NPOS / 32) {
            if (tid < 64) __builtin_prefetch(&ab[(long)tid * NPOS + m0 + 64], 0, 1);
            if (tid < 32)
                tdm_load_2d(lds_off_of(&As[nxt][0][0]), vb + m0 + 32, /*ds=*/1,
                            NPOS, 128, /*tile*/32, 128, NPOS, 1, 3, 3);
            for (int e = tid; e < 512; e += 256) {
                const int n = e >> 3, c4 = e & 7;
                v4f f = *(const v4f*)&ab[(long)n * NPOS + m0 + 32 + c4 * 4];
                v4bf bv;
                bv[0] = (__bf16)f.x; bv[1] = (__bf16)f.y;
                bv[2] = (__bf16)f.z; bv[3] = (__bf16)f.w;
                *(v4bf*)&Bs[nxt][n][c4 * 4] = bv;
            }
        }

        v16bf a[2], bfrag[2];
#pragma unroll
        for (int i = 0; i < 2; i++) {
            // bf16 A 16x32: lanes 0-15 K={0..7,16..23}, lanes 16-31 K={8..15,24..31}
            const int d = db + i * 16 + lm;
            v8bf lo  = *(const v8bf*)&As[cur][d][hi * 8];
            v8bf hi8 = *(const v8bf*)&As[cur][d][16 + hi * 8];
            a[i] = __builtin_shufflevector(lo, hi8, 0, 1, 2, 3, 4, 5, 6, 7,
                                           8, 9, 10, 11, 12, 13, 14, 15);
        }
#pragma unroll
        for (int j = 0; j < 2; j++) {
            // bf16 B 32x16: lanes 0-15 K={0..7,16..23}, lanes 16-31 K={8..15,24..31} (col = lm)
            const int n = nb2 + j * 16 + lm;
            v8bf lo  = *(const v8bf*)&Bs[cur][n][hi * 8];
            v8bf hi8 = *(const v8bf*)&Bs[cur][n][16 + hi * 8];
            bfrag[j] = __builtin_shufflevector(lo, hi8, 0, 1, 2, 3, 4, 5, 6, 7,
                                               8, 9, 10, 11, 12, 13, 14, 15);
        }
#pragma unroll
        for (int i = 0; i < 2; i++)
#pragma unroll
            for (int j = 0; j < 2; j++)
                acc[i][j] = __builtin_amdgcn_wmma_f32_16x16x32_bf16(
                    false, a[i], false, bfrag[j], (short)0, acc[i][j], false, false);
        // no trailing barrier: next iteration writes only the opposite buffers,
        // and its leading barrier orders the Bs[nxt] stores with the reads.
    }

    const float g = gammap[0];
#pragma unroll
    for (int i = 0; i < 2; i++)
#pragma unroll
        for (int j = 0; j < 2; j++)
#pragma unroll
            for (int r = 0; r < 8; r++) {
                const int d = d0 + db + i * 16 + hi * 8 + r;
                const int n = n0 + nb2 + j * 16 + lm;
                const long idx = (long)(b * DV + d) * NPOS + n;  // == x layout
                out[idx] = g * acc[i][j][r] + x[idx];
            }
}

// ---------------------------------------------------------------------------
extern "C" void kernel_launch(void* const* d_in, const int* in_sizes, int n_in,
                              void* d_out, int out_size, void* d_ws, size_t ws_size,
                              hipStream_t stream) {
    const float* x     = (const float*)d_in[0];
    const float* Wq    = (const float*)d_in[1];
    const float* bq    = (const float*)d_in[2];
    const float* Wk    = (const float*)d_in[3];
    const float* bk    = (const float*)d_in[4];
    const float* Wv    = (const float*)d_in[5];
    const float* bv    = (const float*)d_in[6];
    const float* gamma = (const float*)d_in[7];

    float* out = (float*)d_out;
    float* att = out + (long)BATCH * CCH * TT * NPOS;  // second tuple output

    float*  qws = (float*)d_ws;
    float*  kws = qws + (long)BATCH * DQK * NPOS;
    __bf16* vws = (__bf16*)(kws + (long)BATCH * DQK * NPOS);

    proj_kernel<<<(BATCH * TT * NPOS) / 256, 256, 0, stream>>>(
        x, Wq, bq, Wk, bk, Wv, bv, qws, kws, vws);
    energy_kernel<<<dim3(NPOS / 64, NPOS / 128, BATCH), 256, 0, stream>>>(
        qws, kws, att);
    softmax_kernel<<<BATCH * NPOS, 256, 0, stream>>>(att);
    av_kernel<<<dim3(NPOS / 64, DV / 128, BATCH), 256, 0, stream>>>(
        vws, att, x, gamma, out);
}
